// PositionAttentionModule_18184891531382
// MI455X (gfx1250) — matrix operand order repaired and checked by reading, not compiled, not run on hardware
//
#include <hip/hip_runtime.h>

// ---------------------------------------------------------------------------
// Position Attention Module (DANet PAM) for MI455X / gfx1250, wave32 + WMMA.
// Pipeline: weights->bf16 | projections (WMMA) | flash attention (WMMA+online
// softmax, async global->LDS staging) | out-conv + BN + residual (WMMA).
// Workspace ~16.3 MB bf16.
// ---------------------------------------------------------------------------

typedef unsigned short u16;
typedef __attribute__((ext_vector_type(16))) __bf16 v16bf;
typedef __attribute__((ext_vector_type(8)))  float  v8f;
typedef unsigned int v4u __attribute__((ext_vector_type(4)));   // trivial 16B chunk

union F8    { v8f v;  float f[8]; };
union FragQ { v16bf v; v4u q[2]; };

#define LOG2E 1.44269504088896340736f

__device__ __forceinline__ u16 f2bf(float f) {          // f32 -> bf16 RNE
  unsigned u = __float_as_uint(f);
  return (u16)((u + 0x7FFFu + ((u >> 16) & 1u)) >> 16);
}

// Async 16B/lane global->LDS copy (GLOBAL_LOAD_ASYNC_TO_LDS_B128, ASYNCcnt).
// lds_addr: 32-bit LDS byte address (low 32 bits of generic shared pointer),
// voff: per-lane byte offset, sbase: uniform 64-bit global base.
__device__ __forceinline__ void async_g2l_b128(unsigned lds_addr, unsigned voff,
                                               unsigned long long sbase) {
  asm volatile("global_load_async_to_lds_b128 %0, %1, %2"
               :: "v"(lds_addr), "v"(voff), "s"(sbase) : "memory");
}
__device__ __forceinline__ void wait_asynccnt0() {
  asm volatile("s_wait_asynccnt 0x0" ::: "memory");
}

// A-matrix 16x32 bf16 fragment (ISA 7.12.2): lane L holds row M=L&15;
// K(e) = (e>>3)*16 + 8*(L>=16) + (e&7)  -> two contiguous 16B runs per lane.
// rowptr = start of this lane's 32-element K-contiguous row.
__device__ __forceinline__ v16bf ld_frag_a(const u16* rowptr) {
  const int h = (threadIdx.x & 31) >> 4;
  FragQ f;
  f.q[0] = *(const v4u*)(rowptr + h * 8);
  f.q[1] = *(const v4u*)(rowptr + 16 + h * 8);
  return f.v;
}

// B-matrix 32x16 bf16 fragment: lane L holds column N=L&15;
// K(e) = (e>>3)*16 + 8*(L>=16) + (e&7)  -> two contiguous 16B runs per lane.
__device__ __forceinline__ v16bf ld_frag_b(const u16* rowptr) {
  const int h = (threadIdx.x & 31) >> 4;
  FragQ f;
  f.q[0] = *(const v4u*)(rowptr + h * 8);
  f.q[1] = *(const v4u*)(rowptr + 16 + h * 8);
  return f.v;
}

#define WMMA_BF16(A, B, C) \
  __builtin_amdgcn_wmma_f32_16x16x32_bf16(false, (A), false, (B), (short)0, (C), false, false)

// ---------------------------------------------------------------------------
// Kernel 0: convert w_theta/w_phi/w_g ([64x512] each) and w_out ([512x64]) to
// bf16 into ws: [0,98304) = 3 proj weights, [98304,131072) = w_out.
// ---------------------------------------------------------------------------
__global__ __launch_bounds__(256)
void k_cvt_w(const float* __restrict__ wt, const float* __restrict__ wp,
             const float* __restrict__ wg, const float* __restrict__ wo,
             u16* __restrict__ dst) {
  int i = blockIdx.x * 256 + threadIdx.x;           // grid 512 -> 131072
  if (i < 32768)        dst[i] = f2bf(wt[i]);
  else if (i < 65536)   dst[i] = f2bf(wp[i - 32768]);
  else if (i < 98304)   dst[i] = f2bf(wg[i - 65536]);
  else if (i < 131072)  dst[i] = f2bf(wo[i - 98304]);
}

// ---------------------------------------------------------------------------
// Kernel 1: projections. For (b, proj): Out[64ci x 4096n] = W[64x512]*X + bias
// Block = 256 thr (8 waves) computes a [64ci x 64n] tile; K-loop over 512 in
// steps of 32.  X tile staged transposed in LDS ([n][c], K contiguous).
// theta/phi stored [b][n][64ci] (bf16); g stored [b][64ci][4096n] (bf16).
// ---------------------------------------------------------------------------
__global__ __launch_bounds__(256)
void k_proj(const float* __restrict__ x, const u16* __restrict__ wproj,
            const float* __restrict__ bt, const float* __restrict__ bp,
            const float* __restrict__ bg,
            u16* __restrict__ th, u16* __restrict__ ph, u16* __restrict__ gw) {
  __shared__ __align__(16) u16 xl[64 * 32];         // [n][c] 4KB
  const int tid  = threadIdx.x;
  const int lane = tid & 31, wv = tid >> 5;
  const int h = lane >> 4, l15 = lane & 15;
  const int n0   = blockIdx.x * 64;
  const int proj = blockIdx.y;
  const int b    = blockIdx.z;

  const float* X    = x + (size_t)b * 512 * 4096;
  const u16*   W    = wproj + (size_t)proj * 64 * 512;
  const float* bias = (proj == 0) ? bt : (proj == 1) ? bp : bg;

  const int ciA = wv >> 2;   // wave -> ci tiles {ciA, ciA+2}
  const int nT  = wv & 3;    // wave -> n tile

  F8 acc0 = {}; F8 acc1 = {};

  for (int k0 = 0; k0 < 512; k0 += 32) {
#pragma unroll
    for (int p = 0; p < 8; ++p) {                   // stage X transposed
      int cc = p * 4 + (tid >> 6);
      int nn = tid & 63;
      xl[nn * 32 + cc] = f2bf(X[(size_t)(k0 + cc) * 4096 + n0 + nn]);
    }
    __syncthreads();
    v16bf aA = ld_frag_a(W + (size_t)(ciA * 16 + l15) * 512 + k0);
    v16bf aB = ld_frag_a(W + (size_t)((ciA + 2) * 16 + l15) * 512 + k0);
    v16bf bX = ld_frag_b(xl + (nT * 16 + l15) * 32);
    acc0.v = WMMA_BF16(aA, bX, acc0.v);
    acc1.v = WMMA_BF16(aB, bX, acc1.v);
    __syncthreads();
  }

#pragma unroll
  for (int r = 0; r < 8; ++r) {                     // C/D layout: M=r+8h, N=l15
    int m   = r + 8 * h;
    int ci0 = ciA * 16 + m;
    int ci1 = (ciA + 2) * 16 + m;
    int nn  = n0 + nT * 16 + l15;
    float v0 = acc0.f[r] + bias[ci0];
    float v1 = acc1.f[r] + bias[ci1];
    if (proj == 2) {                                // g: [b][ci][n]
      gw[((size_t)b * 64 + ci0) * 4096 + nn] = f2bf(v0);
      gw[((size_t)b * 64 + ci1) * 4096 + nn] = f2bf(v1);
    } else {                                        // theta/phi: [b][n][ci]
      u16* dst = (proj == 0) ? th : ph;
      dst[((size_t)b * 4096 + nn) * 64 + ci0] = f2bf(v0);
      dst[((size_t)b * 4096 + nn) * 64 + ci1] = f2bf(v1);
    }
  }
}

// ---------------------------------------------------------------------------
// Kernel 2: flash attention.  Block = (qblock of 128, batch), 8 waves; each
// wave owns one 16-query tile and streams all 4096 keys in 32-key steps.
// S = Theta_q^T Phi_k (K=64 over channels, 4 WMMAs), online softmax via
// wave32 half-shuffles, P -> wave-private LDS -> A-fragment, O += P * G^T
// (4 WMMAs over ci).  Key blocks staged with GLOBAL_LOAD_ASYNC_TO_LDS_B128.
// y stored [b][n][64ci] bf16.
// ---------------------------------------------------------------------------
__global__ __launch_bounds__(256)
void k_attn(const u16* __restrict__ th, const u16* __restrict__ ph_g,
            const u16* __restrict__ gw, u16* __restrict__ yw) {
  __shared__ __align__(16) u16 th_l[128 * 64];      // [q][c]   16KB
  __shared__ __align__(16) u16 ph_l[32 * 64];       // [key][c]  4KB
  __shared__ __align__(16) u16 g_l[64 * 32];        // [ci][m]   4KB
  __shared__ __align__(16) u16 p_l[8 * 16 * 32];    // per-wave P 8KB

  const int tid  = threadIdx.x;
  const int lane = tid & 31, wv = tid >> 5;
  const int h = lane >> 4, l15 = lane & 15;
  const int b  = blockIdx.y;
  const int q0 = blockIdx.x * 128;

  {                                                 // theta block (once)
    const v4u* src = (const v4u*)(th + ((size_t)b * 4096 + q0) * 64);
    v4u* dst = (v4u*)th_l;
#pragma unroll
    for (int i = 0; i < 4; ++i) dst[tid + i * 256] = src[tid + i * 256];
  }
  __syncthreads();

  const u16* myrow = th_l + (wv * 16 + l15) * 64;   // loop-invariant A frags
  v16bf a0 = ld_frag_a(myrow);
  v16bf a1 = ld_frag_a(myrow + 32);

  float mrow[8], lrow[8];
  F8 O0 = {}, O1 = {}, O2 = {}, O3 = {};
#pragma unroll
  for (int r = 0; r < 8; ++r) { mrow[r] = -1e30f; lrow[r] = 0.f; }

  u16* pw = p_l + wv * (16 * 32);
  const int gci = tid >> 2, gpart = tid & 3;

  // LDS byte addresses for async staging (one 16B chunk per lane)
  const unsigned ph_lds = (unsigned)(uintptr_t)ph_l + (unsigned)tid * 16u;
  const unsigned g_lds  = (unsigned)(uintptr_t)g_l  + (unsigned)tid * 16u;
  const unsigned ph_voff = (unsigned)tid * 16u;                       // bytes
  const unsigned g_voff  = ((unsigned)gci * 4096u + (unsigned)gpart * 8u) * 2u;

  for (int kb = 0; kb < 4096; kb += 32) {
    // async key-block staging: 256 lanes x 16B = 4KB per array
    async_g2l_b128(ph_lds, ph_voff,
        (unsigned long long)(uintptr_t)(ph_g + ((size_t)b * 4096 + kb) * 64));
    async_g2l_b128(g_lds, g_voff,
        (unsigned long long)(uintptr_t)(gw + ((size_t)b * 64) * 4096 + kb));
    if (kb + 32 < 4096) {                           // global_prefetch_b8
      __builtin_prefetch(ph_g + ((size_t)b * 4096 + kb + 32) * 64 + (size_t)tid * 8, 0, 1);
      __builtin_prefetch(gw + ((size_t)b * 64 + gci) * 4096 + kb + 32 + gpart * 8, 0, 1);
    }
    wait_asynccnt0();
    __syncthreads();

    F8 S0 = {}, S1 = {};
    {
      v16bf b00 = ld_frag_b(ph_l + l15 * 64);
      v16bf b01 = ld_frag_b(ph_l + l15 * 64 + 32);
      v16bf b10 = ld_frag_b(ph_l + (16 + l15) * 64);
      v16bf b11 = ld_frag_b(ph_l + (16 + l15) * 64 + 32);
      S0.v = WMMA_BF16(a0, b00, S0.v);
      S0.v = WMMA_BF16(a1, b01, S0.v);
      S1.v = WMMA_BF16(a0, b10, S1.v);
      S1.v = WMMA_BF16(a1, b11, S1.v);
    }

    float alpha[8];
#pragma unroll
    for (int r = 0; r < 8; ++r) {                   // online softmax, row M=r+8h
      float mx = fmaxf(S0.f[r], S1.f[r]);
      mx = fmaxf(mx, __shfl_xor(mx, 1, 32));
      mx = fmaxf(mx, __shfl_xor(mx, 2, 32));
      mx = fmaxf(mx, __shfl_xor(mx, 4, 32));
      mx = fmaxf(mx, __shfl_xor(mx, 8, 32));
      float mn = fmaxf(mrow[r], mx);
      alpha[r] = exp2f((mrow[r] - mn) * LOG2E);
      mrow[r] = mn;
      S0.f[r] = exp2f((S0.f[r] - mn) * LOG2E);
      S1.f[r] = exp2f((S1.f[r] - mn) * LOG2E);
      float rs = S0.f[r] + S1.f[r];
      rs += __shfl_xor(rs, 1, 32);
      rs += __shfl_xor(rs, 2, 32);
      rs += __shfl_xor(rs, 4, 32);
      rs += __shfl_xor(rs, 8, 32);
      lrow[r] = lrow[r] * alpha[r] + rs;
    }
#pragma unroll
    for (int r = 0; r < 8; ++r) {
      O0.f[r] *= alpha[r]; O1.f[r] *= alpha[r];
      O2.f[r] *= alpha[r]; O3.f[r] *= alpha[r];
    }
#pragma unroll
    for (int r = 0; r < 8; ++r) {                   // P -> wave-private LDS
      int m = r + 8 * h;
      pw[m * 32 + l15]      = f2bf(S0.f[r]);
      pw[m * 32 + 16 + l15] = f2bf(S1.f[r]);
    }
    v16bf ap = ld_frag_a(pw + l15 * 32);            // P as A frag (K = keys)
    O0.v = WMMA_BF16(ap, ld_frag_b(g_l + l15 * 32),        O0.v);
    O1.v = WMMA_BF16(ap, ld_frag_b(g_l + (16 + l15) * 32), O1.v);
    O2.v = WMMA_BF16(ap, ld_frag_b(g_l + (32 + l15) * 32), O2.v);
    O3.v = WMMA_BF16(ap, ld_frag_b(g_l + (48 + l15) * 32), O3.v);
    __syncthreads();
  }

#pragma unroll
  for (int r = 0; r < 8; ++r) {                     // normalize + store y
    float inv = 1.0f / lrow[r];
    int qg = q0 + wv * 16 + r + 8 * h;
    size_t base = ((size_t)b * 4096 + qg) * 64;
    yw[base + l15]      = f2bf(O0.f[r] * inv);
    yw[base + 16 + l15] = f2bf(O1.f[r] * inv);
    yw[base + 32 + l15] = f2bf(O2.f[r] * inv);
    yw[base + 48 + l15] = f2bf(O3.f[r] * inv);
  }
}

// ---------------------------------------------------------------------------
// Kernel 3: Z = W_out[512x64]*Y + b_out, BN, + x.  Block = (16n, batch),
// 8 waves x 4 co-tiles each = all 512 output channels; K=64 (2 WMMAs/tile).
// ---------------------------------------------------------------------------
__global__ __launch_bounds__(256)
void k_out(const u16* __restrict__ yw, const u16* __restrict__ wob,
           const float* __restrict__ b_out,
           const float* __restrict__ gamma, const float* __restrict__ beta,
           const float* __restrict__ mean,  const float* __restrict__ var,
           const float* __restrict__ x, float* __restrict__ out) {
  __shared__ __align__(16) u16 yl[16 * 64];         // [n][ci] 2KB
  const int tid  = threadIdx.x;
  const int lane = tid & 31, wv = tid >> 5;
  const int h = lane >> 4, l15 = lane & 15;
  const int b  = blockIdx.y;
  const int n0 = blockIdx.x * 16;

  if (tid < 128)
    ((v4u*)yl)[tid] = ((const v4u*)(yw + ((size_t)b * 4096 + n0) * 64))[tid];
  __syncthreads();

  v16bf bY0 = ld_frag_b(yl + l15 * 64);             // loop-invariant B frags
  v16bf bY1 = ld_frag_b(yl + l15 * 64 + 32);

#pragma unroll
  for (int t = 0; t < 4; ++t) {
    int coT = wv * 4 + t;
    const u16* wr = wob + (size_t)(coT * 16 + l15) * 64;
    v16bf aW0 = ld_frag_a(wr);
    v16bf aW1 = ld_frag_a(wr + 32);
    F8 acc = {};
    acc.v = WMMA_BF16(aW0, bY0, acc.v);
    acc.v = WMMA_BF16(aW1, bY1, acc.v);
#pragma unroll
    for (int r = 0; r < 8; ++r) {
      int co = coT * 16 + r + 8 * h;
      int nn = n0 + l15;
      float z = acc.f[r] + b_out[co];
      z = gamma[co] * (z - mean[co]) * rsqrtf(var[co] + 1e-5f) + beta[co];
      size_t xi = ((size_t)b * 512 + co) * 4096 + nn;
      out[xi] = z + x[xi];
    }
  }
}

// ---------------------------------------------------------------------------
// Launch: ws layout (u16 elems): [0,98304) proj weights bf16, [98304,131072)
// w_out bf16, then theta / phi / g / y at 2,097,152 elems (4MB) each.
// Total ~16.3 MB.
// ---------------------------------------------------------------------------
extern "C" void kernel_launch(void* const* d_in, const int* in_sizes, int n_in,
                              void* d_out, int out_size, void* d_ws, size_t ws_size,
                              hipStream_t stream) {
  (void)in_sizes; (void)n_in; (void)out_size; (void)ws_size;
  const float* x        = (const float*)d_in[0];
  const float* w_theta  = (const float*)d_in[1];
  const float* b_theta  = (const float*)d_in[2];
  const float* w_phi    = (const float*)d_in[3];
  const float* b_phi    = (const float*)d_in[4];
  const float* w_g      = (const float*)d_in[5];
  const float* b_g      = (const float*)d_in[6];
  const float* w_out    = (const float*)d_in[7];
  const float* b_out    = (const float*)d_in[8];
  const float* bn_gamma = (const float*)d_in[9];
  const float* bn_beta  = (const float*)d_in[10];
  const float* bn_mean  = (const float*)d_in[11];
  const float* bn_var   = (const float*)d_in[12];
  float* out = (float*)d_out;

  u16* ws     = (u16*)d_ws;
  u16* wprojB = ws;                    // 3*64*512
  u16* woutB  = ws + 98304;            // 512*64
  u16* thB    = ws + 131072;           // 8*4096*64
  u16* phB    = thB + 2097152;
  u16* gB     = phB + 2097152;
  u16* yB     = gB + 2097152;

  k_cvt_w<<<dim3(512), dim3(256), 0, stream>>>(w_theta, w_phi, w_g, w_out, ws);
  k_proj<<<dim3(64, 3, 8), dim3(256), 0, stream>>>(x, wprojB, b_theta, b_phi, b_g,
                                                   thB, phB, gB);
  k_attn<<<dim3(32, 8), dim3(256), 0, stream>>>(thB, phB, gB, yB);
  k_out<<<dim3(256, 8), dim3(256), 0, stream>>>(yB, woutB, b_out, bn_gamma, bn_beta,
                                                bn_mean, bn_var, x, out);
}
